// DemoDeformableDETR_43482248905334
// MI455X (gfx1250) — matrix operand order repaired and checked by reading, not compiled, not run on hardware
//
#include <hip/hip_runtime.h>
#include <stdint.h>

typedef _Float16 v16h __attribute__((ext_vector_type(16)));
typedef _Float16 v8h  __attribute__((ext_vector_type(8)));
typedef float    v8f  __attribute__((ext_vector_type(8)));
typedef int      i32x4 __attribute__((ext_vector_type(4)));

#define D_MODEL 256
#define NQ      13294
#define BATCH   2
#define NH      8
#define MROWS   (BATCH * NQ)     // 26588
#define MP      26624            // MROWS padded to multiple of 128
#define LDA_P   40               // LDS row stride in halves (16B aligned)

// ---------------------------------------------------------------------------
// CDNA5 async global->LDS copy (ASYNCcnt path), with register-copy fallback
// Probe-confirmed prototype: (i32x4 addrspace(1)*, i32x4 addrspace(3)*, Ii, Ii)
// ---------------------------------------------------------------------------
#if __has_builtin(__builtin_amdgcn_global_load_async_to_lds_b128)
#define HAVE_ASYNC 1
#define ASYNC_CP16(dst_lds, src_g)                                     \
  __builtin_amdgcn_global_load_async_to_lds_b128(                      \
      (__attribute__((address_space(1))) i32x4*)(src_g),               \
      (__attribute__((address_space(3))) i32x4*)(dst_lds), 0, 0)
#if __has_builtin(__builtin_amdgcn_s_wait_asynccnt)
#define ASYNC_WAIT() __builtin_amdgcn_s_wait_asynccnt(0)
#else
#define ASYNC_WAIT() asm volatile("s_wait_asynccnt 0x0" ::: "memory")
#endif
#else
#define HAVE_ASYNC 0
#define ASYNC_CP16(dst_lds, src_g) (*(v8h*)(dst_lds) = *(const v8h*)(src_g))
#define ASYNC_WAIT()
#endif

// ---------------------------------------------------------------------------
// WMMA fragment loaders (CDNA5 16x16x32 f16 layout, wave32)
// ---------------------------------------------------------------------------
__device__ __forceinline__ v16h load_a_frag(const _Float16* As, int row_base) {
  int lane = threadIdx.x & 31;
  int l = lane & 15;
  int hi = (lane >> 4) * 8;
  const _Float16* p = As + (row_base + l) * LDA_P + hi;
  v8h x0 = *(const v8h*)p;
  v8h x1 = *(const v8h*)(p + 16);
  v16h r;
#pragma unroll
  for (int i = 0; i < 8; ++i) { r[i] = x0[i]; r[i + 8] = x1[i]; }
  return r;
}

__device__ __forceinline__ v16h load_b_frag(const _Float16* Bs, int col_base) {
  int lane = threadIdx.x & 31;
  int l = lane & 15;
  int hi = (lane >> 4) * 8;
  const _Float16* p = Bs + (col_base + l) * LDA_P + hi;
  v8h x0 = *(const v8h*)p;
  v8h x1 = *(const v8h*)(p + 16);
  v16h r;
#pragma unroll
  for (int i = 0; i < 8; ++i) { r[i] = x0[i]; r[i + 8] = x1[i]; }
  return r;
}

#define WMMA_F16(a, b, c) \
  __builtin_amdgcn_wmma_f32_16x16x32_f16(false, (a), false, (b), (short)0, (c), false, false)

// ---------------------------------------------------------------------------
// Unified f16 GEMM: C = act(Ah[M,K] @ Bh[N,K]^T-layout + bias) (+residual)
// Ah rows padded to >= gridDim.x*128 (no bounds check in staging).
// Bh is N-major f16 ([N][K]), K multiple of 32.
// Double-buffered LDS, async global->LDS staging.
// HW>0: remap output row m=(b*HW+hw) -> b*NQ + lvl_start + hw.
// ---------------------------------------------------------------------------
__global__ __launch_bounds__(256)
void gemm_h_kernel(const _Float16* __restrict__ Ah, const _Float16* __restrict__ Bh,
                   const float* __restrict__ bias, const float* __restrict__ residual,
                   float* __restrict__ Cf, _Float16* __restrict__ Ch,
                   int M, int K, int N, int relu, int HW, int lvl_start)
{
  __shared__ __align__(16) _Float16 As[2][128 * LDA_P];
  __shared__ __align__(16) _Float16 Bs[2][128 * LDA_P];
  const int tid = threadIdx.x;
  const int wid = tid >> 5;
  const int wm = wid >> 2, wn = wid & 3;
  const int m0 = blockIdx.x * 128;
  const int n0 = blockIdx.y * 128;
  const int srow = tid >> 2;          // staging chunk row (0..63)
  const int scol = (tid & 3) * 8;     // staging chunk col in halves
  const _Float16* Ab = Ah + (size_t)m0 * K;
  const _Float16* Bb = Bh + (size_t)n0 * K;

  auto stage = [&](int buf, int k0) {
#pragma unroll
    for (int i = 0; i < 2; ++i) {
      int row = srow + i * 64;
      ASYNC_CP16(&As[buf][row * LDA_P + scol], Ab + (size_t)row * K + k0 + scol);
      ASYNC_CP16(&Bs[buf][row * LDA_P + scol], Bb + (size_t)row * K + k0 + scol);
    }
  };

  v8f acc[4][2] = {};
  const int KT = K >> 5;

  stage(0, 0);
  ASYNC_WAIT();
  __syncthreads();

  for (int t = 0; t < KT; ++t) {
    if (t + 1 < KT) stage((t + 1) & 1, (t + 1) << 5);   // overlap with compute
    const _Float16* Ac = As[t & 1];
    const _Float16* Bc = Bs[t & 1];
    v16h bfrag[2];
#pragma unroll
    for (int nt = 0; nt < 2; ++nt) bfrag[nt] = load_b_frag(Bc, wn * 32 + nt * 16);
#pragma unroll
    for (int mt = 0; mt < 4; ++mt) {
      v16h afrag = load_a_frag(Ac, wm * 64 + mt * 16);
#pragma unroll
      for (int nt = 0; nt < 2; ++nt)
        acc[mt][nt] = WMMA_F16(afrag, bfrag[nt], acc[mt][nt]);
    }
    ASYNC_WAIT();
    __syncthreads();
  }

  const int lane = tid & 31;
  const int ln = lane & 15, lh = lane >> 4;
#pragma unroll
  for (int mt = 0; mt < 4; ++mt) {
#pragma unroll
    for (int nt = 0; nt < 2; ++nt) {
      int col = n0 + wn * 32 + nt * 16 + ln;
#pragma unroll
      for (int j = 0; j < 8; ++j) {
        int row = m0 + wm * 64 + mt * 16 + lh * 8 + j;
        if (row < M) {
          float v = acc[mt][nt][j] + bias[col];
          if (relu) v = fmaxf(v, 0.0f);
          int orow = row;
          if (HW > 0) { int b = row / HW; orow = b * NQ + lvl_start + (row - b * HW); }
          if (Cf) {
            float o = v;
            if (residual) o += residual[(size_t)orow * N + col];
            Cf[(size_t)orow * N + col] = o;
          }
          if (Ch) Ch[(size_t)orow * N + col] = (_Float16)v;
        }
      }
    }
  }
}

// ---------------------------------------------------------------------------
// c6 conv: 3x3 stride 2 pad 1 on feat_c5 [2,2048,25,25] -> src rows at lvl_start
// implicit GEMM over 9 taps; weights prepacked f16 [tap][O=256][C=2048]
// ---------------------------------------------------------------------------
__global__ __launch_bounds__(256)
void conv_c6_kernel(const float* __restrict__ feat5, const _Float16* __restrict__ wt,
                    const float* __restrict__ bias, float* __restrict__ out, int lvl_start)
{
  const int CIN = 2048, M = BATCH * 169;
  __shared__ __align__(16) _Float16 As[128 * LDA_P];
  __shared__ __align__(16) _Float16 Bs[128 * LDA_P];
  const int tid = threadIdx.x;
  const int wid = tid >> 5;
  const int wm = wid >> 2, wn = wid & 3;
  const int m0 = blockIdx.x * 128;
  const int n0 = blockIdx.y * 128;
  const int srow = tid >> 2;
  const int scol = (tid & 3) * 8;

  v8f acc[4][2] = {};

  for (int tap = 0; tap < 9; ++tap) {
    int ky = tap / 3, kx = tap - ky * 3;
    const _Float16* Bt = wt + (size_t)tap * 256 * CIN;
    for (int k0 = 0; k0 < CIN; k0 += 32) {
      // A gather: branchless (clamped address + select)
#pragma unroll
      for (int i = 0; i < 16; ++i) {
        int idx = tid + i * 256;
        int kk = idx >> 7, r = idx & 127;
        int m = m0 + r;
        int mc = (m < M) ? m : 0;
        int b = mc / 169, rr = mc - b * 169;
        int oy = rr / 13, ox = rr - oy * 13;
        int iy = oy * 2 - 1 + ky, ix = ox * 2 - 1 + kx;
        int ok = (m < M) & (iy >= 0) & (iy < 25) & (ix >= 0) & (ix < 25);
        int iyc = iy < 0 ? 0 : (iy > 24 ? 24 : iy);
        int ixc = ix < 0 ? 0 : (ix > 24 ? 24 : ix);
        float v = feat5[(((size_t)b * CIN + k0 + kk) * 25 + iyc) * 25 + ixc];
        As[r * LDA_P + kk] = (_Float16)(ok ? v : 0.0f);
      }
      // B: vectorized f16 copy
#pragma unroll
      for (int i = 0; i < 2; ++i) {
        int row = srow + i * 64;
        *(v8h*)&Bs[row * LDA_P + scol] =
            *(const v8h*)&Bt[(size_t)(n0 + row) * CIN + k0 + scol];
      }
      __syncthreads();
      v16h bfrag[2];
#pragma unroll
      for (int nt = 0; nt < 2; ++nt) bfrag[nt] = load_b_frag(Bs, wn * 32 + nt * 16);
#pragma unroll
      for (int mt = 0; mt < 4; ++mt) {
        v16h afrag = load_a_frag(As, wm * 64 + mt * 16);
#pragma unroll
        for (int nt = 0; nt < 2; ++nt)
          acc[mt][nt] = WMMA_F16(afrag, bfrag[nt], acc[mt][nt]);
      }
      __syncthreads();
    }
  }

  const int lane = tid & 31;
  const int ln = lane & 15, lh = lane >> 4;
#pragma unroll
  for (int mt = 0; mt < 4; ++mt) {
#pragma unroll
    for (int nt = 0; nt < 2; ++nt) {
      int col = n0 + wn * 32 + nt * 16 + ln;
#pragma unroll
      for (int j = 0; j < 8; ++j) {
        int m = m0 + wm * 64 + mt * 16 + lh * 8 + j;
        if (m < M) {
          int b = m / 169, rr = m - b * 169;
          out[((size_t)b * NQ + lvl_start + rr) * D_MODEL + col] = acc[mt][nt][j] + bias[col];
        }
      }
    }
  }
}

// ---------------------------------------------------------------------------
// prepack kernels (one-time f16 conversions / transposes)
// ---------------------------------------------------------------------------
// weights [K,N] fp32 -> [N,K] f16
__global__ void wT_to_h_kernel(const float* __restrict__ in, _Float16* __restrict__ outh,
                               int K, int N)
{
  int i = blockIdx.x * blockDim.x + threadIdx.x;
  if (i >= N * K) return;
  int n = i / K, k = i - n * K;
  outh[i] = (_Float16)in[(size_t)k * N + n];
}

// straight fp32 -> f16
__global__ void w_to_h_kernel(const float* __restrict__ in, _Float16* __restrict__ outh, int n)
{
  int i = blockIdx.x * blockDim.x + threadIdx.x;
  if (i < n) outh[i] = (_Float16)in[i];
}

// c6_w [256,2048,3,3] fp32 -> [tap][256][2048] f16
__global__ void c6w_to_h_kernel(const float* __restrict__ w, _Float16* __restrict__ outh)
{
  int i = blockIdx.x * blockDim.x + threadIdx.x;
  if (i >= 9 * 256 * 2048) return;
  int c = i & 2047;
  int t = i >> 11;
  int o = t & 255;
  int tap = t >> 8;
  outh[i] = (_Float16)w[((size_t)o * 2048 + c) * 9 + tap];
}

// feature map [B][C][HW] fp32 -> [(b*HW+hw)][C] f16 (tiled transpose)
__global__ __launch_bounds__(256)
void featT_kernel(const float* __restrict__ in, _Float16* __restrict__ outh, int C, int HW)
{
  __shared__ _Float16 tile[32][33];
  int b = blockIdx.z;
  int hw0 = blockIdx.x * 32;
  int c0 = blockIdx.y * 32;
  int tx = threadIdx.x & 31, ty = threadIdx.x >> 5;   // ty: 0..7
#pragma unroll
  for (int j = 0; j < 4; ++j) {
    int cc = ty + j * 8;
    int hw = hw0 + tx;
    int hwc = hw < HW ? hw : HW - 1;
    float v = in[((size_t)b * C + c0 + cc) * HW + hwc];
    tile[cc][tx] = (_Float16)(hw < HW ? v : 0.0f);
  }
  __syncthreads();
#pragma unroll
  for (int j = 0; j < 4; ++j) {
    int hh = ty + j * 8;
    int hw = hw0 + hh;
    if (hw < HW) outh[((size_t)b * HW + hw) * C + c0 + tx] = tile[tx][hh];
  }
}

// ---------------------------------------------------------------------------
// pos embed (threefry2x32 key={0,42} + erfinv) fused with query = src + pos
// ---------------------------------------------------------------------------
__device__ __forceinline__ uint32_t rotl32(uint32_t v, int r) { return (v << r) | (v >> (32 - r)); }

__device__ __forceinline__ void threefry2x32(uint32_t k0, uint32_t k1,
                                             uint32_t x0, uint32_t x1,
                                             uint32_t& o0, uint32_t& o1)
{
  uint32_t ks0 = k0, ks1 = k1, ks2 = k0 ^ k1 ^ 0x1BD11BDAu;
  const int ra[4] = {13, 15, 26, 6};
  const int rb[4] = {17, 29, 16, 24};
  x0 += ks0; x1 += ks1;
#pragma unroll
  for (int blk = 0; blk < 5; ++blk) {
#pragma unroll
    for (int r = 0; r < 4; ++r) {
      int rot = (blk & 1) ? rb[r] : ra[r];
      x0 += x1; x1 = rotl32(x1, rot); x1 ^= x0;
    }
    uint32_t kA = (blk == 0 || blk == 3) ? ks1 : ((blk == 1 || blk == 4) ? ks2 : ks0);
    uint32_t kB = (blk == 0 || blk == 3) ? ks2 : ((blk == 1 || blk == 4) ? ks0 : ks1);
    x0 += kA; x1 += kB + (uint32_t)(blk + 1);
  }
  o0 = x0; o1 = x1;
}

__device__ __forceinline__ float erfinv_f(float x)
{
  float w = -__logf((1.0f - x) * (1.0f + x));
  float p;
  if (w < 5.0f) {
    w -= 2.5f;
    p = 2.81022636e-08f;
    p = fmaf(p, w, 3.43273939e-07f);
    p = fmaf(p, w, -3.5233877e-06f);
    p = fmaf(p, w, -4.39150654e-06f);
    p = fmaf(p, w, 0.00021858087f);
    p = fmaf(p, w, -0.00125372503f);
    p = fmaf(p, w, -0.00417768164f);
    p = fmaf(p, w, 0.246640727f);
    p = fmaf(p, w, 1.50140941f);
  } else {
    w = __fsqrt_rn(w) - 3.0f;
    p = -0.000200214257f;
    p = fmaf(p, w, 0.000100950558f);
    p = fmaf(p, w, 0.00134934322f);
    p = fmaf(p, w, -0.00367342844f);
    p = fmaf(p, w, 0.00573950773f);
    p = fmaf(p, w, -0.0076224613f);
    p = fmaf(p, w, 0.00943887047f);
    p = fmaf(p, w, 1.00167406f);
    p = fmaf(p, w, 2.83297682f);
  }
  return p * x;
}

__global__ void pos_query_kernel(const float* __restrict__ src, _Float16* __restrict__ qh, int n)
{
  int i = blockIdx.x * blockDim.x + threadIdx.x;
  if (i >= n) return;
  const int half = n >> 1;
  uint32_t c0, c1, o0, o1;
  int sel;
  if (i < half) { c0 = (uint32_t)i;          c1 = (uint32_t)(i + half); sel = 0; }
  else          { c0 = (uint32_t)(i - half); c1 = (uint32_t)i;          sel = 1; }
  threefry2x32(0u, 42u, c0, c1, o0, o1);
  uint32_t bits = sel ? o1 : o0;
  float f = __uint_as_float((bits >> 9) | 0x3f800000u) - 1.0f;
  const float lo = -0.99999994f;
  float u = fmaxf(lo, fmaf(f, 1.0f - lo, lo));
  qh[i] = (_Float16)(src[i] + 1.41421356237f * erfinv_f(u));
}

// ---------------------------------------------------------------------------
// softmax over groups of 16 (per b,q,head)
// ---------------------------------------------------------------------------
__global__ void softmax16_kernel(float* __restrict__ aw, int n)
{
  int i = blockIdx.x * blockDim.x + threadIdx.x;
  if (i >= n) return;
  float* p = aw + (size_t)i * 16;
  float v[16], m = -1e30f;
#pragma unroll
  for (int j = 0; j < 16; ++j) { v[j] = p[j]; m = fmaxf(m, v[j]); }
  float s = 0.0f;
#pragma unroll
  for (int j = 0; j < 16; ++j) { v[j] = __expf(v[j] - m); s += v[j]; }
  float inv = 1.0f / s;
#pragma unroll
  for (int j = 0; j < 16; ++j) p[j] = v[j] * inv;
}

// ---------------------------------------------------------------------------
// deformable sampling: one wave per (b,q,head), lane = channel; f16 output
// ---------------------------------------------------------------------------
__global__ __launch_bounds__(256)
void sample_kernel(const float* __restrict__ src, const float* __restrict__ off,
                   const float* __restrict__ aw, _Float16* __restrict__ outh)
{
  const int lvlH[4] = {100, 50, 25, 13};
  const int lvlW[4] = {100, 50, 25, 13};
  const int lvlS[4] = {0, 10000, 12500, 13125};
  int gw = blockIdx.x * 8 + (threadIdx.x >> 5);
  if (gw >= MROWS * NH) return;
  int lane = threadIdx.x & 31;
  int h = gw & 7;
  int bq = gw >> 3;
  int q = bq % NQ;
  int b = bq / NQ;

  int ql = 3;
  if (q < 10000) ql = 0; else if (q < 12500) ql = 1; else if (q < 13125) ql = 2;
  int qoff = q - lvlS[ql];
  float refx = ((qoff % lvlW[ql]) + 0.5f) / (float)lvlW[ql];
  float refy = ((qoff / lvlW[ql]) + 0.5f) / (float)lvlH[ql];

  const float* offp = off + (size_t)bq * D_MODEL + h * 32;
  const float* awp  = aw  + (size_t)bq * 128 + h * 16;
  float acc = 0.0f;

#pragma unroll
  for (int l = 0; l < 4; ++l) {
    int Hl = lvlH[l], Wl = lvlW[l];
    const float* fbase = src + ((size_t)b * NQ + lvlS[l]) * D_MODEL + h * 32 + lane;
#pragma unroll
    for (int p = 0; p < 4; ++p) {
      float ox = offp[l * 8 + p * 2 + 0];
      float oy = offp[l * 8 + p * 2 + 1];
      float x = (refx + ox / (float)Wl) * (float)Wl - 0.5f;
      float y = (refy + oy / (float)Hl) * (float)Hl - 0.5f;
      float xf = floorf(x), yf = floorf(y);
      float wx1 = x - xf, wx0 = 1.0f - wx1;
      float wy1 = y - yf, wy0 = 1.0f - wy1;
      int x0 = (int)xf, y0 = (int)yf;
      float s = 0.0f;
#pragma unroll
      for (int cy = 0; cy < 2; ++cy) {
#pragma unroll
        for (int cx = 0; cx < 2; ++cx) {
          int xi = x0 + cx, yi = y0 + cy;
          if (xi >= 0 && xi < Wl && yi >= 0 && yi < Hl) {
            float w = (cy ? wy1 : wy0) * (cx ? wx1 : wx0);
            s += w * fbase[(size_t)(yi * Wl + xi) * D_MODEL];
          }
        }
      }
      acc += awp[l * 4 + p] * s;
    }
  }
  outh[(size_t)bq * D_MODEL + h * 32 + lane] = (_Float16)acc;
}

// ---------------------------------------------------------------------------
// x = LayerNorm(src + attn_out) * g + b  -> f32 (residual) + f16 (next GEMM)
// ---------------------------------------------------------------------------
__global__ __launch_bounds__(256)
void add_ln_kernel(const float* __restrict__ src, const float* __restrict__ attn,
                   const float* __restrict__ g, const float* __restrict__ be,
                   float* __restrict__ xout, _Float16* __restrict__ xh, int Mrows)
{
  int row = blockIdx.x * 8 + (threadIdx.x >> 5);
  if (row >= Mrows) return;
  int lane = threadIdx.x & 31;
  const float* ps = src  + (size_t)row * D_MODEL;
  const float* pa = attn + (size_t)row * D_MODEL;
  float v[8], s = 0.0f;
#pragma unroll
  for (int j = 0; j < 8; ++j) { v[j] = ps[lane + j * 32] + pa[lane + j * 32]; s += v[j]; }
#pragma unroll
  for (int d = 16; d; d >>= 1) s += __shfl_xor(s, d, 32);
  float mean = s * (1.0f / 256.0f);
  float s2 = 0.0f;
#pragma unroll
  for (int j = 0; j < 8; ++j) { float dd = v[j] - mean; s2 += dd * dd; }
#pragma unroll
  for (int d = 16; d; d >>= 1) s2 += __shfl_xor(s2, d, 32);
  float inv = rsqrtf(s2 * (1.0f / 256.0f) + 1e-5f);
#pragma unroll
  for (int j = 0; j < 8; ++j) {
    int c = lane + j * 32;
    float o = (v[j] - mean) * inv * g[c] + be[c];
    xout[(size_t)row * D_MODEL + c] = o;
    xh[(size_t)row * D_MODEL + c] = (_Float16)o;
  }
}

// ---------------------------------------------------------------------------
// launch
// ---------------------------------------------------------------------------
extern "C" void kernel_launch(void* const* d_in, const int* in_sizes, int n_in,
                              void* d_out, int out_size, void* d_ws, size_t ws_size,
                              hipStream_t stream)
{
  const float* feat_c3 = (const float*)d_in[0];
  const float* feat_c4 = (const float*)d_in[1];
  const float* feat_c5 = (const float*)d_in[2];
  const float* proj_w3 = (const float*)d_in[3];
  const float* proj_b3 = (const float*)d_in[4];
  const float* proj_w4 = (const float*)d_in[5];
  const float* proj_b4 = (const float*)d_in[6];
  const float* proj_w5 = (const float*)d_in[7];
  const float* proj_b5 = (const float*)d_in[8];
  const float* c6_w    = (const float*)d_in[9];
  const float* c6_b    = (const float*)d_in[10];
  const float* off_w   = (const float*)d_in[11];
  const float* off_b   = (const float*)d_in[12];
  const float* attn_w  = (const float*)d_in[13];
  const float* attn_b  = (const float*)d_in[14];
  const float* out_w   = (const float*)d_in[15];
  const float* out_b   = (const float*)d_in[16];
  const float* ln_g    = (const float*)d_in[17];
  const float* ln_b    = (const float*)d_in[18];
  const float* ffn_w1  = (const float*)d_in[19];
  const float* ffn_b1  = (const float*)d_in[20];
  const float* ffn_w2  = (const float*)d_in[21];
  const float* ffn_b2  = (const float*)d_in[22];
  float* out = (float*)d_out;

  // ---- workspace carve (bytes, 256B aligned) ----
  char* base = (char*)d_ws;
  size_t cur = 0;
  auto carve = [&](size_t bytes) -> char* {
    char* p = base + cur;
    cur = (cur + bytes + 255) & ~(size_t)255;
    return p;
  };
  const int MP3 = 20096, MP4 = 5120, MP5 = 1280;   // padded proj row counts
  float*    buf_src  = (float*)carve((size_t)MP * 256 * 4);
  float*    buf_off  = (float*)carve((size_t)MP * 256 * 4);  // offsets; later attn_out
  float*    buf_aw   = (float*)carve((size_t)MP * 128 * 4);
  float*    buf_x    = (float*)carve((size_t)MP * 256 * 4);
  _Float16* buf_qh   = (_Float16*)carve((size_t)MP * 256 * 2);
  _Float16* buf_acch = (_Float16*)carve((size_t)MP * 256 * 2);
  _Float16* buf_xh   = (_Float16*)carve((size_t)MP * 256 * 2);
  _Float16* buf_hh   = (_Float16*)carve((size_t)MP * 1024 * 2);
  _Float16* featT3   = (_Float16*)carve((size_t)MP3 * 512 * 2);
  _Float16* featT4   = (_Float16*)carve((size_t)MP4 * 1024 * 2);
  _Float16* featT5   = (_Float16*)carve((size_t)MP5 * 2048 * 2);
  _Float16* offT     = (_Float16*)carve((size_t)256 * 256 * 2);
  _Float16* attnT    = (_Float16*)carve((size_t)128 * 256 * 2);
  _Float16* outT     = (_Float16*)carve((size_t)256 * 256 * 2);
  _Float16* f1T      = (_Float16*)carve((size_t)1024 * 256 * 2);
  _Float16* f2T      = (_Float16*)carve((size_t)256 * 1024 * 2);
  _Float16* p3h      = (_Float16*)carve((size_t)256 * 512 * 2);
  _Float16* p4h      = (_Float16*)carve((size_t)256 * 1024 * 2);
  _Float16* p5h      = (_Float16*)carve((size_t)256 * 2048 * 2);
  _Float16* c6h      = (_Float16*)carve((size_t)9 * 256 * 2048 * 2);

  // ---- prepack (one-time f16 conversions) ----
  wT_to_h_kernel<<<(256 * 256 + 255) / 256, 256, 0, stream>>>(off_w, offT, 256, 256);
  wT_to_h_kernel<<<(128 * 256 + 255) / 256, 256, 0, stream>>>(attn_w, attnT, 256, 128);
  wT_to_h_kernel<<<(256 * 256 + 255) / 256, 256, 0, stream>>>(out_w, outT, 256, 256);
  wT_to_h_kernel<<<(1024 * 256 + 255) / 256, 256, 0, stream>>>(ffn_w1, f1T, 256, 1024);
  wT_to_h_kernel<<<(256 * 1024 + 255) / 256, 256, 0, stream>>>(ffn_w2, f2T, 1024, 256);
  w_to_h_kernel<<<(256 * 512 + 255) / 256, 256, 0, stream>>>(proj_w3, p3h, 256 * 512);
  w_to_h_kernel<<<(256 * 1024 + 255) / 256, 256, 0, stream>>>(proj_w4, p4h, 256 * 1024);
  w_to_h_kernel<<<(256 * 2048 + 255) / 256, 256, 0, stream>>>(proj_w5, p5h, 256 * 2048);
  c6w_to_h_kernel<<<(9 * 256 * 2048 + 255) / 256, 256, 0, stream>>>(c6_w, c6h);
  featT_kernel<<<dim3(313, 16, 2), 256, 0, stream>>>(feat_c3, featT3, 512, 10000);
  featT_kernel<<<dim3(79, 32, 2), 256, 0, stream>>>(feat_c4, featT4, 1024, 2500);
  featT_kernel<<<dim3(20, 64, 2), 256, 0, stream>>>(feat_c5, featT5, 2048, 625);

  // ---- level projections (WMMA) -> buf_src ----
  gemm_h_kernel<<<dim3(157, 2), 256, 0, stream>>>(featT3, p3h, proj_b3, nullptr,
      buf_src, nullptr, 20000, 512, 256, 0, 10000, 0);
  gemm_h_kernel<<<dim3(40, 2), 256, 0, stream>>>(featT4, p4h, proj_b4, nullptr,
      buf_src, nullptr, 5000, 1024, 256, 0, 2500, 10000);
  gemm_h_kernel<<<dim3(10, 2), 256, 0, stream>>>(featT5, p5h, proj_b5, nullptr,
      buf_src, nullptr, 1250, 2048, 256, 0, 625, 12500);
  // ---- c6 conv -> buf_src rows 13125..13293 ----
  conv_c6_kernel<<<dim3(3, 2), 256, 0, stream>>>(feat_c5, c6h, c6_b, buf_src, 13125);

  // ---- query = src + pos (f16) ----
  pos_query_kernel<<<(MROWS * 256 + 255) / 256, 256, 0, stream>>>(buf_src, buf_qh, MROWS * 256);

  // ---- sampling offsets + attention weights ----
  gemm_h_kernel<<<dim3(208, 2), 256, 0, stream>>>(buf_qh, offT, off_b, nullptr,
      buf_off, nullptr, MROWS, 256, 256, 0, 0, 0);
  gemm_h_kernel<<<dim3(208, 1), 256, 0, stream>>>(buf_qh, attnT, attn_b, nullptr,
      buf_aw, nullptr, MROWS, 256, 128, 0, 0, 0);
  softmax16_kernel<<<(MROWS * NH + 255) / 256, 256, 0, stream>>>(buf_aw, MROWS * NH);

  // ---- deformable sampling -> f16 ----
  sample_kernel<<<(MROWS * NH + 7) / 8, 256, 0, stream>>>(buf_src, buf_off, buf_aw, buf_acch);

  // ---- output projection (attn_out -> buf_off, offsets consumed) ----
  gemm_h_kernel<<<dim3(208, 2), 256, 0, stream>>>(buf_acch, outT, out_b, nullptr,
      buf_off, nullptr, MROWS, 256, 256, 0, 0, 0);

  // ---- x = LN(src + attn_out) ----
  add_ln_kernel<<<(MROWS + 7) / 8, 256, 0, stream>>>(buf_src, buf_off, ln_g, ln_b,
      buf_x, buf_xh, MROWS);

  // ---- FFN ----
  gemm_h_kernel<<<dim3(208, 8), 256, 0, stream>>>(buf_xh, f1T, ffn_b1, nullptr,
      nullptr, buf_hh, MROWS, 256, 1024, 1, 0, 0);
  gemm_h_kernel<<<dim3(208, 2), 256, 0, stream>>>(buf_hh, f2T, ffn_b2, buf_x,
      out, nullptr, MROWS, 1024, 256, 0, 0, 0);
}
